// GraphMETNetwork_75977971466535
// MI455X (gfx1250) — compile-verified, matrix operand order checked by reading, not run on hardware
//
#include <hip/hip_runtime.h>
#include <hip/hip_bf16.h>
#include <hip/hip_fp16.h>

typedef __attribute__((ext_vector_type(16))) _Float16 v16h;
typedef __attribute__((ext_vector_type(8)))  float    v8f;

__device__ __forceinline__ float elu1(float x) { return x > 0.0f ? x : expm1f(x); }

// Monotone float -> uint key so unsigned atomicMax == float max.
__device__ __forceinline__ unsigned fkey(float f) {
  unsigned u = __float_as_uint(f);
  return (u & 0x80000000u) ? ~u : (u | 0x80000000u);
}
__device__ __forceinline__ float fdec(unsigned k) {
  unsigned u = (k & 0x80000000u) ? (k ^ 0x80000000u) : ~k;
  return __uint_as_float(u);
}

// ---------------------------------------------------------------------------
// K0: zero agg keys + stat accumulators, convert W_msg (64x32 f32) -> f16
// ---------------------------------------------------------------------------
__global__ void k_init(unsigned* __restrict__ keys, long long nk,
                       float* __restrict__ scal,
                       const float* __restrict__ wmsg, _Float16* __restrict__ wh) {
  long long idx = (long long)blockIdx.x * blockDim.x + threadIdx.x;
  long long stride = (long long)gridDim.x * blockDim.x;
  for (long long i = idx; i < nk; i += stride) keys[i] = 0u;
  if (idx < 256) scal[idx] = 0.0f;
  if (idx < 64 * 32) wh[idx] = (_Float16)wmsg[idx];
}

// ---------------------------------------------------------------------------
// K1: per-node embedding: emb_cont / emb_cat / encoder, pre-BN "z" rows,
//     plus per-channel sum / sum-of-squares via LDS reduce + global atomics.
// ---------------------------------------------------------------------------
__global__ void k_node(const float* __restrict__ xc, const int* __restrict__ xcat,
                       const float* __restrict__ Wc,   const float* __restrict__ bc,
                       const float* __restrict__ Tch,  const float* __restrict__ Tpd,
                       const float* __restrict__ Tpv,
                       const float* __restrict__ Wcat, const float* __restrict__ bcat,
                       const float* __restrict__ Wenc, const float* __restrict__ benc,
                       float* __restrict__ z, float* __restrict__ sum1,
                       float* __restrict__ sq1, int N) {
  __shared__ float sS[32], sQ[32];
  if (threadIdx.x < 32) { sS[threadIdx.x] = 0.0f; sQ[threadIdx.x] = 0.0f; }
  __syncthreads();

  int i = blockIdx.x * blockDim.x + threadIdx.x;
  if (i < N) {
    float x[6];
#pragma unroll
    for (int k = 0; k < 6; ++k) x[k] = xc[(size_t)i * 6 + k];
    float f[32];
#pragma unroll
    for (int j = 0; j < 16; ++j) {
      float s = bc[j];
#pragma unroll
      for (int k = 0; k < 6; ++k) s += x[k] * Wc[k * 16 + j];
      f[16 + j] = elu1(s);                       // emb_cont -> features 16..31
    }
    int c0 = xcat[(size_t)i * 3 + 0];
    int c1 = xcat[(size_t)i * 3 + 1];
    int c2 = xcat[(size_t)i * 3 + 2];
    int p = c0 < 0 ? -c0 : c0;
    const int pdgs[7] = {1, 2, 11, 13, 22, 130, 211};
    int pi = p;
#pragma unroll
    for (int t = 0; t < 7; ++t) if (p == pdgs[t]) pi = t;
    float cat[24];
#pragma unroll
    for (int k = 0; k < 8; ++k) {
      cat[k]      = Tch[(c1 + 1) * 8 + k];
      cat[8 + k]  = Tpd[pi * 8 + k];
      cat[16 + k] = Tpv[c2 * 8 + k];
    }
#pragma unroll
    for (int j = 0; j < 16; ++j) {
      float s = bcat[j];
#pragma unroll
      for (int k = 0; k < 24; ++k) s += cat[k] * Wcat[k * 16 + j];
      f[j] = elu1(s);                            // emb_cat -> features 0..15
    }
#pragma unroll
    for (int j = 0; j < 32; ++j) {
      float s = benc[j];
#pragma unroll
      for (int k = 0; k < 32; ++k) s += f[k] * Wenc[k * 32 + j];
      float v = elu1(s);
      z[(size_t)i * 32 + j] = v;
      atomicAdd(&sS[j], v);
      atomicAdd(&sQ[j], v * v);
    }
  }
  __syncthreads();
  if (threadIdx.x < 32) {
    atomicAdd(&sum1[threadIdx.x], sS[threadIdx.x]);
    atomicAdd(&sq1[threadIdx.x],  sQ[threadIdx.x]);
  }
}

// ---------------------------------------------------------------------------
// K2/K6: fold batchnorm into affine y = a*x + b
// ---------------------------------------------------------------------------
__global__ void k_bnfin(const float* __restrict__ sum, const float* __restrict__ sq,
                        const float* __restrict__ g, const float* __restrict__ be,
                        float* __restrict__ a, float* __restrict__ b, float invN) {
  int c = threadIdx.x;
  if (c < 32) {
    float mu  = sum[c] * invN;
    float var = sq[c] * invN - mu * mu;
    float rs  = rsqrtf(var + 1e-5f);
    float aa  = g[c] * rs;
    a[c] = aa;
    b[c] = be[c] - mu * aa;
  }
}

// ---------------------------------------------------------------------------
// K3: apply BN1; materialize normalized emb in f32 (for residual/output) and
//     f16 (WMMA operand feed for the edge kernel).
// ---------------------------------------------------------------------------
__global__ void k_applybn(const float* __restrict__ z, const float* __restrict__ a,
                          const float* __restrict__ b, float* __restrict__ ef,
                          _Float16* __restrict__ eh, long long total) {
  long long idx = (long long)blockIdx.x * blockDim.x + threadIdx.x;
  if (idx < total) {
    int c = (int)(idx & 31);
    float v = z[idx] * a[c] + b[c];
    ef[idx] = v;
    eh[idx] = (_Float16)v;
  }
}

// ---------------------------------------------------------------------------
// K4: edge message GEMM on WMMA + segment-max scatter.
//     One wave = 16 edges (M=16). K=64 = [x_i | x_j - x_i] as two K=32 f16
//     chunks; N=32 = two 16-column tiles. 4x v_wmma_f32_16x16x32_f16 / tile.
//     B fragments (W_msg) loop-invariant; dst indices shared via __shfl;
//     full tiles take a branch-free scatter fast path.
// ---------------------------------------------------------------------------
__global__ void k_edge(const int* __restrict__ ei, const _Float16* __restrict__ embh,
                       const _Float16* __restrict__ wh, const float* __restrict__ bmsg,
                       unsigned* __restrict__ aggk, int E, int ntiles) {
  const int lane = threadIdx.x & 31;
  const int m  = lane & 15;     // row (A) / column (B,C,D) within 16
  const int hf = lane >> 4;     // half-wave select for the 16-bit K striping
  const int kb0 = hf * 8;       // v16h elems [0..7]  hold K = kb0..kb0+7
  const int kb1 = 16 + hf * 8;  // v16h elems [8..15] hold K = kb1..kb1+7

  v16h B00, B01, B10, B11;      // [col-tile][K-chunk]
#pragma unroll
  for (int i = 0; i < 8; ++i) {
    B00[i]     = wh[(kb0 + i) * 32 + m];
    B00[8 + i] = wh[(kb1 + i) * 32 + m];
    B01[i]     = wh[(32 + kb0 + i) * 32 + m];
    B01[8 + i] = wh[(32 + kb1 + i) * 32 + m];
    B10[i]     = wh[(kb0 + i) * 32 + 16 + m];
    B10[8 + i] = wh[(kb1 + i) * 32 + 16 + m];
    B11[i]     = wh[(32 + kb0 + i) * 32 + 16 + m];
    B11[8 + i] = wh[(32 + kb1 + i) * 32 + 16 + m];
  }
  const float bias0 = bmsg[m];
  const float bias1 = bmsg[16 + m];

  const int* __restrict__ srcp = ei;
  const int* __restrict__ dstp = ei + E;

  int wave = (int)((blockIdx.x * blockDim.x + threadIdx.x) >> 5);
  int nw   = (int)(((long long)gridDim.x * blockDim.x) >> 5);

  for (int tile = wave; tile < ntiles; tile += nw) {
    int e  = tile * 16 + m;
    int ec = e < E ? e : E - 1;              // clamped gather (tail tiles)
    int sn = srcp[ec], dn = dstp[ec];
    const _Float16* sr = embh + (size_t)sn * 32;
    const _Float16* dr = embh + (size_t)dn * 32;

    v16h A0, A1;                              // chunk0 = x_i, chunk1 = x_j - x_i
#pragma unroll
    for (int i = 0; i < 8; ++i) {
      _Float16 d0 = dr[kb0 + i], d1 = dr[kb1 + i];
      _Float16 s0 = sr[kb0 + i], s1 = sr[kb1 + i];
      A0[i] = d0;               A0[8 + i] = d1;
      A1[i] = (_Float16)(s0 - d0); A1[8 + i] = (_Float16)(s1 - d1);
    }
    v8f C0, C1;
#pragma unroll
    for (int r = 0; r < 8; ++r) { C0[r] = bias0; C1[r] = bias1; }

    C0 = __builtin_amdgcn_wmma_f32_16x16x32_f16(false, A0, false, B00, (short)0, C0, false, false);
    C0 = __builtin_amdgcn_wmma_f32_16x16x32_f16(false, A1, false, B01, (short)0, C0, false, false);
    C1 = __builtin_amdgcn_wmma_f32_16x16x32_f16(false, A0, false, B10, (short)0, C1, false, false);
    C1 = __builtin_amdgcn_wmma_f32_16x16x32_f16(false, A1, false, B11, (short)0, C1, false, false);

    // D layout: VGPR r holds row (r + hf*8), column = m (+16 for tile 1).
    // Row (hf*8 + r)'s dst index lives in lane (hf*8 + r)'s `dn` -> __shfl,
    // issued in uniform control flow (valid for tail tiles too).
    if (tile * 16 + 16 <= E) {               // full tile: branch-free scatter
#pragma unroll
      for (int r = 0; r < 8; ++r) {
        int node = __shfl(dn, hf * 8 + r, 32);
        atomicMax(&aggk[(size_t)node * 32 + m],      fkey(C0[r]));
        atomicMax(&aggk[(size_t)node * 32 + 16 + m], fkey(C1[r]));
      }
    } else {                                  // tail tile: per-row guard
#pragma unroll
      for (int r = 0; r < 8; ++r) {
        int node = __shfl(dn, hf * 8 + r, 32);
        int ee = tile * 16 + hf * 8 + r;
        if (ee < E) {
          atomicMax(&aggk[(size_t)node * 32 + m],      fkey(C0[r]));
          atomicMax(&aggk[(size_t)node * 32 + 16 + m], fkey(C1[r]));
        }
      }
    }
  }
}

// ---------------------------------------------------------------------------
// K5: decode max-keys -> agg f32 (empty segment / non-finite -> 0) + BN2 stats
// ---------------------------------------------------------------------------
__global__ void k_aggred(const unsigned* __restrict__ keys, float* __restrict__ agg,
                         float* __restrict__ sum2, float* __restrict__ sq2,
                         long long total) {
  __shared__ float sS[32], sQ[32];
  if (threadIdx.x < 32) { sS[threadIdx.x] = 0.0f; sQ[threadIdx.x] = 0.0f; }
  __syncthreads();
  long long idx = (long long)blockIdx.x * blockDim.x + threadIdx.x;
  if (idx < total) {
    unsigned k = keys[idx];
    float v = 0.0f;
    if (k != 0u) {
      v = fdec(k);
      if (!__builtin_isfinite(v)) v = 0.0f;
    }
    agg[idx] = v;
    int c = (int)(idx & 31);
    atomicAdd(&sS[c], v);
    atomicAdd(&sQ[c], v * v);
  }
  __syncthreads();
  if (threadIdx.x < 32) {
    atomicAdd(&sum2[threadIdx.x], sS[threadIdx.x]);
    atomicAdd(&sq2[threadIdx.x],  sQ[threadIdx.x]);
  }
}

// ---------------------------------------------------------------------------
// K7: residual + output MLP (32 -> elu 16 -> 1)
// ---------------------------------------------------------------------------
__global__ void k_out(const float* __restrict__ ef, const float* __restrict__ agg,
                      const float* __restrict__ a2, const float* __restrict__ b2,
                      const float* __restrict__ Wo1, const float* __restrict__ bo1,
                      const float* __restrict__ Wo2, const float* __restrict__ bo2,
                      float* __restrict__ out, int N) {
  int i = blockIdx.x * blockDim.x + threadIdx.x;
  if (i >= N) return;
  float h[32];
#pragma unroll
  for (int c = 0; c < 32; ++c)
    h[c] = ef[(size_t)i * 32 + c] + agg[(size_t)i * 32 + c] * a2[c] + b2[c];
  float o = bo2[0];
#pragma unroll
  for (int j = 0; j < 16; ++j) {
    float s = bo1[j];
#pragma unroll
    for (int c = 0; c < 32; ++c) s += h[c] * Wo1[c * 16 + j];
    o += elu1(s) * Wo2[j];
  }
  out[i] = o;
}

extern "C" void kernel_launch(void* const* d_in, const int* in_sizes, int n_in,
                              void* d_out, int out_size, void* d_ws, size_t ws_size,
                              hipStream_t stream) {
  (void)n_in; (void)out_size; (void)ws_size;
  const float* x_cont  = (const float*)d_in[0];
  const int*   x_cat   = (const int*)d_in[1];
  const int*   ei      = (const int*)d_in[2];
  /* d_in[3] batch: single graph, unused */
  const float* Wc      = (const float*)d_in[4];
  const float* bc      = (const float*)d_in[5];
  const float* Tch     = (const float*)d_in[6];
  const float* Tpd     = (const float*)d_in[7];
  const float* Tpv     = (const float*)d_in[8];
  const float* Wcat    = (const float*)d_in[9];
  const float* bcat    = (const float*)d_in[10];
  const float* Wenc    = (const float*)d_in[11];
  const float* benc    = (const float*)d_in[12];
  const float* g_all   = (const float*)d_in[13];
  const float* be_all  = (const float*)d_in[14];
  const float* Wmsg    = (const float*)d_in[15];
  const float* bmsg    = (const float*)d_in[16];
  const float* g_conv  = (const float*)d_in[17];
  const float* be_conv = (const float*)d_in[18];
  const float* Wo1     = (const float*)d_in[19];
  const float* bo1     = (const float*)d_in[20];
  const float* Wo2     = (const float*)d_in[21];
  const float* bo2     = (const float*)d_in[22];
  float* out = (float*)d_out;

  const int N = in_sizes[0] / 6;
  const int E = in_sizes[2] / 2;
  const long long total = (long long)N * 32;

  char* ws = (char*)d_ws;
  float*    zbuf = (float*)ws;                         // N*32 f32 (z, reused as agg)
  float*    ef   = (float*)(ws + total * 4);           // N*32 f32 normalized emb
  _Float16* eh   = (_Float16*)(ws + total * 8);        // N*32 f16 normalized emb
  unsigned* keys = (unsigned*)(ws + total * 10);       // N*32 u32 max-keys
  _Float16* wh   = (_Float16*)(ws + total * 14);       // 64*32 f16 W_msg
  float*    scal = (float*)(ws + total * 14 + 4096);   // 256 f32 stats/affines
  float *sum1 = scal,        *sq1 = scal + 32,  *a1 = scal + 64,  *b1s = scal + 96;
  float *sum2 = scal + 128,  *sq2 = scal + 160, *a2 = scal + 192, *b2s = scal + 224;

  const int T = 256;
  const int nblkN = (N + T - 1) / T;
  const int nblkTot = (int)((total + T - 1) / T);
  const int ntiles = (E + 15) / 16;

  k_init<<<1024, T, 0, stream>>>(keys, total, scal, Wmsg, wh);
  k_node<<<nblkN, T, 0, stream>>>(x_cont, x_cat, Wc, bc, Tch, Tpd, Tpv,
                                  Wcat, bcat, Wenc, benc, zbuf, sum1, sq1, N);
  k_bnfin<<<1, 32, 0, stream>>>(sum1, sq1, g_all, be_all, a1, b1s, 1.0f / (float)N);
  k_applybn<<<nblkTot, T, 0, stream>>>(zbuf, a1, b1s, ef, eh, total);
  k_edge<<<2048, T, 0, stream>>>(ei, eh, wh, bmsg, keys, E, ntiles);
  k_aggred<<<nblkTot, T, 0, stream>>>(keys, zbuf, sum2, sq2, total);
  k_bnfin<<<1, 32, 0, stream>>>(sum2, sq2, g_conv, be_conv, a2, b2s, 1.0f / (float)N);
  k_out<<<nblkN, T, 0, stream>>>(ef, zbuf, a2, b2s, Wo1, bo1, Wo2, bo2, out, N);
}
